// BahdanauAttention_34789235098111
// MI455X (gfx1250) — matrix operand order repaired and checked by reading, not compiled, not run on hardware
//
#include <hip/hip_runtime.h>
#include <math.h>

typedef __attribute__((ext_vector_type(16))) __bf16 v16bf;
typedef __attribute__((ext_vector_type(8)))  __bf16 v8bf;
typedef __attribute__((ext_vector_type(8)))  float  v8f;
typedef __attribute__((ext_vector_type(4)))  float  v4f;
typedef __attribute__((ext_vector_type(4)))  unsigned int v4u;
typedef __attribute__((ext_vector_type(8)))  int    v8i;
typedef __attribute__((ext_vector_type(4)))  int    v4i;

#define HIDDEN  512
#define ENCDIM  512
#define BATCH   256
#define SEQ     1024
#define LDB_PAD 520   // 512 bf16 + 16B pad per row (TDM pad: interval 256 DW, amount 4 DW)

#if __has_builtin(__builtin_amdgcn_tensor_load_to_lds) && __has_builtin(__builtin_amdgcn_s_wait_tensorcnt)
#define HAVE_TDM 1
#else
#define HAVE_TDM 0
#endif

// ---------------------------------------------------------------------------
// f32 -> bf16 convert (x4 vectorized; n multiple of 1024)
// ---------------------------------------------------------------------------
__global__ __launch_bounds__(256) void cvt_f32_bf16(const float* __restrict__ src,
                                                    __bf16* __restrict__ dst, int n) {
    int i = (blockIdx.x * blockDim.x + threadIdx.x) * 4;
    if (i + 3 < n) {
        v4f f = *(const v4f*)(src + i);
        dst[i + 0] = (__bf16)f.x;
        dst[i + 1] = (__bf16)f.y;
        dst[i + 2] = (__bf16)f.z;
        dst[i + 3] = (__bf16)f.w;
    }
}

// 16-element bf16 fragment from two 16-byte runs (LDS or global)
__device__ __forceinline__ v16bf ldfrag(const __bf16* p0, const __bf16* p1) {
    v8bf lo = *(const v8bf*)p0;
    v8bf hi = *(const v8bf*)p1;
    return __builtin_shufflevector(lo, hi, 0,1,2,3,4,5,6,7,8,9,10,11,12,13,14,15);
}

// tanh via one v_exp_f32: exact at +/-1 saturation, monotone in between
__device__ __forceinline__ float fast_tanh(float x) {
    float e = __expf(2.0f * x);
    return 1.0f - 2.0f / (e + 1.0f);
}

#if HAVE_TDM
// TDM: load one 16x512 bf16 tile (row stride 512 elems) into LDS with
// per-row padding of 4 DWORDs -> LDS row stride 520 bf16. D# per ISA ch.8.
__device__ __forceinline__ void tdm_load_tile(unsigned int lds_addr, const __bf16* gptr) {
    unsigned long long ga = (unsigned long long)(uintptr_t)gptr;
    v4u g0 = { 1u,                                   // [1:0] count=1 (valid user D#)
               lds_addr,                             // [63:32] lds_addr
               (unsigned int)ga,                     // [95:64] global_addr lo
               ((unsigned int)(ga >> 32) & 0x1FFFFFFu) | (2u << 30) }; // addr hi | type=2
    v8i g1 = { (int)((1u << 16)      // data_size = 1 -> 2 bytes
             |       (1u << 20)      // pad_enable
             |       (7u << 22)      // pad_interval: 256 DWORDs (one 1024B row)
             |       (3u << 25)),    // pad_amount: 4 DWORDs (16B)
               (int)(512u << 16),    // tensor_dim0 = 512 (bits 79:48)
               (int)(16u  << 16),    // tensor_dim0 hi=0 | tensor_dim1 = 16
               (int)(512u << 16),    // tensor_dim1 hi=0 | tile_dim0 = 512
               16,                   // tile_dim1 = 16, tile_dim2 = 0
               512,                  // tensor_dim0_stride = 512 elems
               0, 0 };
    v4i z4 = {0, 0, 0, 0};
#if defined(__clang_major__) && (__clang_major__ >= 23)
    v8i z8 = {0, 0, 0, 0, 0, 0, 0, 0};
    __builtin_amdgcn_tensor_load_to_lds(g0, g1, z4, z4, z8, 0);
#else
    __builtin_amdgcn_tensor_load_to_lds(g0, g1, z4, z4, 0);
#endif
}
#endif

// ---------------------------------------------------------------------------
// proj_h = decoder_hidden @ W_h^T (256x512x512) bf16 WMMA
// ---------------------------------------------------------------------------
__global__ __launch_bounds__(256) void projh_gemm(const __bf16* __restrict__ A,
                                                  const __bf16* __restrict__ Bm,
                                                  float* __restrict__ C) {
    const int tid  = threadIdx.x;
    const int lane = tid & 31;
    const int wave = tid >> 5;
    const int tile = blockIdx.x * 8 + wave;   // 16 Mtiles * 32 Ntiles
    const int mt   = tile >> 5;
    const int nt   = tile & 31;
    const int ln   = lane & 15;
    const int half = lane >> 4;
    const int m = mt * 16 + ln;
    const int n = nt * 16 + ln;

    v8f acc = {};
    for (int k0 = 0; k0 < HIDDEN; k0 += 32) {
        v16bf af = ldfrag(A  + m * HIDDEN + k0 + half * 8,
                          A  + m * HIDDEN + k0 + 16 + half * 8);
        v16bf bf = ldfrag(Bm + n * HIDDEN + k0 + half * 8,
                          Bm + n * HIDDEN + k0 + 16 + half * 8);
        acc = __builtin_amdgcn_wmma_f32_16x16x32_bf16(false, af, false, bf,
                                                      (short)0, acc, false, false);
    }
    float* out = C + (size_t)(mt * 16 + half * 8) * HIDDEN + nt * 16 + ln;
#pragma unroll
    for (int r = 0; r < 8; ++r) out[(size_t)r * HIDDEN] = acc[r];
}

// ---------------------------------------------------------------------------
// Fused: scores = v . tanh(proj_h + enc@W_s^T); softmax; context
// 1 block / batch row; 8 waves. A fragments live in registers (128 VGPRs),
// B tiles double-buffered in LDS via TDM prefetch (wave 0 issues, TENSORcnt).
// ---------------------------------------------------------------------------
__global__ __launch_bounds__(256) void attn_fused(const float*  __restrict__ enc,
                                                  const __bf16* __restrict__ Wsb,
                                                  const float*  __restrict__ projh,
                                                  const float*  __restrict__ vvec,
                                                  float* __restrict__ ctx_out,
                                                  float* __restrict__ alpha_out) {
    __shared__ __bf16 Btile[2][16 * LDB_PAD];   // 2 x 16.6 KB
    __shared__ float  projh_s[HIDDEN];
    __shared__ float  v_s[HIDDEN];
    __shared__ float  scores_s[SEQ];
    __shared__ float  red_s[256];

    const int b    = blockIdx.x;
    const int tid  = threadIdx.x;
    const int lane = tid & 31;
    const int wave = tid >> 5;
    const int ln   = lane & 15;
    const int half = lane >> 4;
    const float* encB = enc + (size_t)b * SEQ * ENCDIM;

    for (int i = tid; i < HIDDEN; i += 256) { projh_s[i] = projh[b * HIDDEN + i]; v_s[i] = vvec[i]; }

#if HAVE_TDM
    const unsigned int ldsB0 = (unsigned int)(uintptr_t)&Btile[0][0];
    const unsigned int ldsB1 = (unsigned int)(uintptr_t)&Btile[1][0];
#endif

#pragma unroll 1
    for (int ablk = 0; ablk < 8; ++ablk) {
        const int mt = ablk * 8 + wave;           // this wave's M-tile (l rows)
#if HAVE_TDM
        if (wave == 0) tdm_load_tile(ldsB0, Wsb); // preload tile nt=0 (buffers free here)
#endif
        // ---- load A fragments for all 16 k-steps into registers (f32 -> bf16) ----
        const float* arow = encB + (size_t)(mt * 16 + ln) * ENCDIM;
        v16bf af[16];
#pragma unroll
        for (int kk = 0; kk < 16; ++kk) {
            const int k0 = kk * 32;
            v4f f0 = *(const v4f*)(arow + k0 + half * 8);
            v4f f1 = *(const v4f*)(arow + k0 + half * 8 + 4);
            v4f f2 = *(const v4f*)(arow + k0 + 16 + half * 8);
            v4f f3 = *(const v4f*)(arow + k0 + 16 + half * 8 + 4);
            v16bf t;
            t[0]  = (__bf16)f0.x; t[1]  = (__bf16)f0.y; t[2]  = (__bf16)f0.z; t[3]  = (__bf16)f0.w;
            t[4]  = (__bf16)f1.x; t[5]  = (__bf16)f1.y; t[6]  = (__bf16)f1.z; t[7]  = (__bf16)f1.w;
            t[8]  = (__bf16)f2.x; t[9]  = (__bf16)f2.y; t[10] = (__bf16)f2.z; t[11] = (__bf16)f2.w;
            t[12] = (__bf16)f3.x; t[13] = (__bf16)f3.y; t[14] = (__bf16)f3.z; t[15] = (__bf16)f3.w;
            af[kk] = t;
        }

        float sacc[8] = {0.f, 0.f, 0.f, 0.f, 0.f, 0.f, 0.f, 0.f};
#pragma unroll 1
        for (int nt = 0; nt < 32; ++nt) {
            __syncthreads();   // everyone done reading buf[(nt+1)&1] (used last iter)
#if HAVE_TDM
            if (wave == 0) {
                if (nt + 1 < 32) {
                    tdm_load_tile((nt & 1) ? ldsB0 : ldsB1,
                                  Wsb + (size_t)(nt + 1) * 16 * HIDDEN);
                    __builtin_amdgcn_s_wait_tensorcnt(1);  // tile nt landed
                } else {
                    __builtin_amdgcn_s_wait_tensorcnt(0);
                }
            }
#else
            {   // synchronous cooperative copy of tile nt
                const __bf16* gsrc = Wsb + (size_t)nt * 16 * HIDDEN;
                __bf16* dstB = &Btile[nt & 1][0];
                for (int c = tid; c < 1024; c += 256) {
                    int row = c >> 6, col = (c & 63) * 8;
                    *(v8bf*)(dstB + row * LDB_PAD + col) =
                        *(const v8bf*)(gsrc + row * HIDDEN + col);
                }
            }
#endif
            __syncthreads();   // tile nt visible to all waves

            const __bf16* Bt = &Btile[nt & 1][0];
            v8f acc = {};
#pragma unroll
            for (int kk = 0; kk < 16; ++kk) {
                const int k0 = kk * 32;
                v16bf bf = ldfrag(Bt + ln * LDB_PAD + k0 + half * 8,
                                  Bt + ln * LDB_PAD + k0 + 16 + half * 8);
                acc = __builtin_amdgcn_wmma_f32_16x16x32_bf16(false, af[kk], false, bf,
                                                              (short)0, acc, false, false);
            }
            // epilogue: lane&15 = h column, VGPR r = l row
            const int   h  = nt * 16 + ln;
            const float ph = projh_s[h];
            const float vh = v_s[h];
#pragma unroll
            for (int r = 0; r < 8; ++r)
                sacc[r] += fast_tanh(ph + acc[r]) * vh;
        }
        // reduce 16 N-lanes inside each half-wave; wave owns its M-tile scores
#pragma unroll
        for (int r = 0; r < 8; ++r) {
            float x = sacc[r];
            x += __shfl_xor(x, 1, 32);
            x += __shfl_xor(x, 2, 32);
            x += __shfl_xor(x, 4, 32);
            x += __shfl_xor(x, 8, 32);
            if (ln == 0) scores_s[mt * 16 + half * 8 + r] = x;
        }
    }
    __syncthreads();

    // ---- softmax over scores_s[0..1023] ----
    float lm = -INFINITY;
    for (int i = tid; i < SEQ; i += 256) lm = fmaxf(lm, scores_s[i]);
    red_s[tid] = lm; __syncthreads();
    for (int s = 128; s > 0; s >>= 1) {
        if (tid < s) red_s[tid] = fmaxf(red_s[tid], red_s[tid + s]);
        __syncthreads();
    }
    const float maxv = red_s[0]; __syncthreads();

    float lsum = 0.f;
    for (int i = tid; i < SEQ; i += 256) {
        float e = __expf(scores_s[i] - maxv);
        scores_s[i] = e;
        lsum += e;
    }
    red_s[tid] = lsum; __syncthreads();
    for (int s = 128; s > 0; s >>= 1) {
        if (tid < s) red_s[tid] += red_s[tid + s];
        __syncthreads();
    }
    const float inv = 1.0f / red_s[0]; __syncthreads();

    for (int i = tid; i < SEQ; i += 256) {
        float a = scores_s[i] * inv;
        scores_s[i] = a;
        alpha_out[(size_t)b * SEQ + i] = a;
    }
    __syncthreads();

    // ---- context[b,e] = sum_l alpha[l] * enc[b,l,e] ----
    const int e0 = tid * 2;
    float cx = 0.f, cy = 0.f;
    for (int l = 0; l < SEQ; ++l) {
        float a = scores_s[l];
        const float2 ev = *(const float2*)(encB + (size_t)l * ENCDIM + e0);
        cx += a * ev.x;
        cy += a * ev.y;
    }
    float2 outv; outv.x = cx; outv.y = cy;
    *(float2*)(ctx_out + (size_t)b * ENCDIM + e0) = outv;
}

// ---------------------------------------------------------------------------
// Host launcher
// ---------------------------------------------------------------------------
extern "C" void kernel_launch(void* const* d_in, const int* in_sizes, int n_in,
                              void* d_out, int out_size, void* d_ws, size_t ws_size,
                              hipStream_t stream) {
    (void)in_sizes; (void)n_in; (void)out_size; (void)ws_size;
    const float* dec = (const float*)d_in[0];
    const float* enc = (const float*)d_in[1];
    const float* W_h = (const float*)d_in[2];
    const float* W_s = (const float*)d_in[3];
    const float* v   = (const float*)d_in[4];

    float* out   = (float*)d_out;
    float* ctx   = out;                       // 256*512
    float* alpha = out + BATCH * HIDDEN;      // 256*1024

    char*   ws    = (char*)d_ws;
    float*  projh = (float*)ws;                           // 512 KB
    __bf16* Wsb   = (__bf16*)(ws + 524288);               // 512 KB
    __bf16* Whb   = (__bf16*)(ws + 1048576);              // 512 KB
    __bf16* decb  = (__bf16*)(ws + 1572864);              // 256 KB

    cvt_f32_bf16<<<262144 / 1024, 256, 0, stream>>>(W_s, Wsb, 262144);
    cvt_f32_bf16<<<262144 / 1024, 256, 0, stream>>>(W_h, Whb, 262144);
    cvt_f32_bf16<<<131072 / 1024, 256, 0, stream>>>(dec, decb, 131072);

    projh_gemm<<<64, 256, 0, stream>>>(decb, Whb, projh);

    attn_fused<<<BATCH, 256, 0, stream>>>(enc, Wsb, projh, v, ctx, alpha);
}
